// Module_84963043049771
// MI455X (gfx1250) — compile-verified, matrix operand order checked by reading, not run on hardware
//
#include <hip/hip_runtime.h>

// Problem sizes (fixed by the reference)
#define BB   64
#define TT   200
#define NIN  128
#define NN   2048
#define NOUT 20
#define KTOT (NN + NIN)   // 2176: [recurrent | input] concatenated K
#define NWG  64           // persistent grid: 4 row-tiles x 16 col-tiles

typedef __attribute__((ext_vector_type(16))) __bf16       v16bf;
typedef __attribute__((ext_vector_type(8)))  float        v8f;
typedef __attribute__((ext_vector_type(4)))  unsigned int u32x4;

union FragU { u32x4 q[2]; v16bf v; };

// ---------------------------------------------------------------- setup ----
__global__ void snn_init_kernel(float* mem, float* wp, __bf16* abuf, int* bar) {
  int i = blockIdx.x * blockDim.x + threadIdx.x;
  int stride = gridDim.x * blockDim.x;
  for (int idx = i; idx < BB * NN; idx += stride) {
    mem[idx] = 0.0f;
    wp[idx]  = 0.0f;
    abuf[idx]           = (__bf16)0.0f;   // a_0 = spike(mem0-1)*(1+wp0) = 0
    abuf[idx + BB * NN] = (__bf16)0.0f;
  }
  if (i == 0) { bar[0] = 0; bar[1] = 0; }
}

// WT[n][k] = Wcomb[k][n]; Wcomb = [ w_signs[r]*|w[r][c]| ; w_in ]  (bf16)
__global__ void snn_build_wt(const float* __restrict__ w,
                             const float* __restrict__ w_in,
                             const float* __restrict__ w_signs,
                             __bf16* __restrict__ wt) {
  int i = blockIdx.x * blockDim.x + threadIdx.x;
  int stride = gridDim.x * blockDim.x;
  const int total = NN * KTOT;
  for (int idx = i; idx < total; idx += stride) {
    int c = idx / KTOT;          // output neuron (column of Wcomb)
    int r = idx % KTOT;          // source row
    float v = (r < NN) ? (w_signs[r] * fabsf(w[(size_t)r * NN + c]))
                       : w_in[(size_t)(r - NN) * NN + c];
    wt[idx] = (__bf16)v;
  }
}

// Xbf[t][b][i] = bf16(inputs[b][t][i])
__global__ void snn_build_x(const float* __restrict__ inputs,
                            __bf16* __restrict__ xbf) {
  int i = blockIdx.x * blockDim.x + threadIdx.x;
  int stride = gridDim.x * blockDim.x;
  const int total = TT * BB * NIN;
  for (int idx = i; idx < total; idx += stride) {
    int t = idx / (BB * NIN);
    int rem = idx % (BB * NIN);
    int b = rem / NIN;
    int c = rem % NIN;
    xbf[idx] = (__bf16)inputs[((size_t)b * TT + t) * NIN + c];
  }
}

// ------------------------------------------------- persistent scan kernel --
__global__ __launch_bounds__(256) void snn_step_kernel(
    const __bf16* __restrict__ WT,    // [NN][KTOT]  B operand (transposed)
    const __bf16* __restrict__ Xbf,   // [TT][BB][NIN]
    __bf16* __restrict__ Abuf,        // [2][BB][NN] double-buffered activations
    float* __restrict__ mem,          // [BB][NN]
    float* __restrict__ wp,           // [BB][NN]
    unsigned char* __restrict__ spk,  // [TT][BB][NN]
    const float* __restrict__ pvec,   // [NN]
    int* __restrict__ bar) {          // [0]=arrive count, [1]=phase
  const int tid  = threadIdx.x;
  const int lane = tid & 31;
  const int wave = tid >> 5;
  const int lm   = lane & 15;
  const int hf   = lane >> 4;        // half-wave select (K staggering)
  const int m0 = (blockIdx.x & 3) * 16;              // batch-row tile
  const int n0 = (blockIdx.x >> 2) * 128 + wave * 16; // neuron-col subtile

  const int m = m0 + lm;                               // A-fragment row (batch)
  const __bf16* btrow = WT + (size_t)(n0 + lm) * KTOT; // B-fragment row

  for (int t = 0; t < TT; ++t) {
    const __bf16* arow = Abuf + (size_t)(t & 1) * (BB * NN) + (size_t)m * NN;
    const __bf16* xrow = Xbf + ((size_t)t * BB + m) * NIN;
    v8f acc = {0.f, 0.f, 0.f, 0.f, 0.f, 0.f, 0.f, 0.f};

    // recurrent part: K = 0..2047   (a_t @ rec_w)
#pragma unroll 4
    for (int kb = 0; kb < NN; kb += 32) {
      FragU af, bfr;
      af.q[0]  = *(const u32x4*)(arow + kb + 8 * hf);
      af.q[1]  = *(const u32x4*)(arow + kb + 16 + 8 * hf);
      bfr.q[0] = *(const u32x4*)(btrow + kb + 8 * hf);
      bfr.q[1] = *(const u32x4*)(btrow + kb + 16 + 8 * hf);
      acc = __builtin_amdgcn_wmma_f32_16x16x32_bf16(
          false, af.v, false, bfr.v, (short)0, acc, false, false);
    }
    // input part: K = 2048..2175    (x_t @ w_in == h1_t, fused)
#pragma unroll
    for (int kb = 0; kb < NIN; kb += 32) {
      FragU af, bfr;
      af.q[0]  = *(const u32x4*)(xrow + kb + 8 * hf);
      af.q[1]  = *(const u32x4*)(xrow + kb + 16 + 8 * hf);
      bfr.q[0] = *(const u32x4*)(btrow + NN + kb + 8 * hf);
      bfr.q[1] = *(const u32x4*)(btrow + NN + kb + 16 + 8 * hf);
      acc = __builtin_amdgcn_wmma_f32_16x16x32_bf16(
          false, af.v, false, bfr.v, (short)0, acc, false, false);
    }

    // fused elementwise state update (each (b,n) owned by exactly one wave)
    __bf16* anext = Abuf + (size_t)((t + 1) & 1) * (BB * NN);
    const int n = n0 + lm;
    const float pv  = pvec[n];
    const float dep = (pv < 0.0f) ? 1.0f : 0.0f;
#pragma unroll
    for (int r = 0; r < 8; ++r) {
      const int b = m0 + r + 8 * hf;          // D layout: vgpr r -> M = r + 8*half
      const size_t idx = (size_t)b * NN + n;
      float mv = mem[idx];
      float wv = wp[idx];
      float outv = (mv - 1.0f) > 0.0f ? 1.0f : 0.0f;   // spike(mem_t - 1)
      spk[((size_t)t * BB + b) * NN + n] = (unsigned char)outv;
      float wpn = wv * 0.99f + outv * pv * (1.0f + dep * wv);
      float mn  = 0.9f * mv + acc[r] - outv;           // syn includes h1_t
      mem[idx] = mn;
      wp[idx]  = wpn;
      float outn = (mn - 1.0f) > 0.0f ? 1.0f : 0.0f;   // next step's spike
      anext[idx] = (__bf16)(outn * (1.0f + wpn));      // next A operand
    }

    // device-wide barrier: one per step (double-buffered A makes this safe)
    __threadfence();
    __syncthreads();
    if (tid == 0) {
      int ph = ((volatile int*)bar)[1];
      if (atomicAdd(bar, 1) == NWG - 1) {
        bar[0] = 0;
        __threadfence();
        ((volatile int*)bar)[1] = ph + 1;
      } else {
        while (((volatile int*)bar)[1] == ph) __builtin_amdgcn_s_sleep(1);
      }
      __threadfence();
    }
    __syncthreads();
  }
}

// ------------------------------------------------------------- readout ----
__global__ __launch_bounds__(256) void snn_out_kernel(
    const unsigned char* __restrict__ spk,
    const float* __restrict__ w_out,   // [NN][NOUT]
    float* __restrict__ out) {         // [BB][TT][NOUT]
  const int b = blockIdx.x;
  const int tid = threadIdx.x;
  const int lane = tid & 31, wave = tid >> 5;
  __shared__ float wred[8][NOUT];
  float oacc = 0.0f;   // leaky output state (valid for tid < NOUT)

  for (int t = 0; t < TT; ++t) {
    float a20[NOUT];
#pragma unroll
    for (int o = 0; o < NOUT; ++o) a20[o] = 0.0f;
    const unsigned char* row = spk + ((size_t)t * BB + b) * NN;
    for (int n = tid; n < NN; n += 256) {
      if (row[n]) {
        const float* wr = w_out + (size_t)n * NOUT;
#pragma unroll
        for (int o = 0; o < NOUT; ++o) a20[o] += wr[o];
      }
    }
#pragma unroll
    for (int o = 0; o < NOUT; ++o) {
      float v = a20[o];
      for (int off = 16; off > 0; off >>= 1) v += __shfl_down(v, off, 32);
      if (lane == 0) wred[wave][o] = v;
    }
    __syncthreads();
    if (tid < NOUT) {
      float s = 0.0f;
#pragma unroll
      for (int w8 = 0; w8 < 8; ++w8) s += wred[w8][tid];
      oacc = 0.9f * oacc + s;                              // ALPHA_MEM_OUT
      out[((size_t)b * TT + t) * NOUT + tid] = oacc;
    }
    __syncthreads();
  }
}

// -------------------------------------------------------------- launcher --
extern "C" void kernel_launch(void* const* d_in, const int* in_sizes, int n_in,
                              void* d_out, int out_size, void* d_ws, size_t ws_size,
                              hipStream_t stream) {
  const float* inputs  = (const float*)d_in[0];
  const float* w       = (const float*)d_in[1];
  const float* w_in    = (const float*)d_in[2];
  const float* w_out   = (const float*)d_in[3];
  const float* w_signs = (const float*)d_in[4];
  const float* p       = (const float*)d_in[5];
  (void)in_sizes; (void)n_in; (void)out_size; (void)ws_size;

  char* ws = (char*)d_ws;
  size_t off = 0;
  auto alloc = [&](size_t bytes) -> char* {
    char* ptr = ws + off;
    off = (off + bytes + 255) & ~(size_t)255;
    return ptr;
  };
  __bf16* WT   = (__bf16*)alloc((size_t)NN * KTOT * 2);   // ~8.9 MB
  __bf16* Xbf  = (__bf16*)alloc((size_t)TT * BB * NIN * 2); // ~3.3 MB
  __bf16* Abuf = (__bf16*)alloc((size_t)2 * BB * NN * 2); // 0.5 MB
  float*  mem  = (float*)alloc((size_t)BB * NN * 4);      // 0.5 MB
  float*  wpb  = (float*)alloc((size_t)BB * NN * 4);      // 0.5 MB
  unsigned char* spk = (unsigned char*)alloc((size_t)TT * BB * NN); // 26 MB
  int* bar = (int*)alloc(256);

  snn_init_kernel<<<128, 256, 0, stream>>>(mem, wpb, Abuf, bar);
  snn_build_wt<<<1024, 256, 0, stream>>>(w, w_in, w_signs, WT);
  snn_build_x<<<512, 256, 0, stream>>>(inputs, Xbf);
  snn_step_kernel<<<NWG, 256, 0, stream>>>(WT, Xbf, Abuf, mem, wpb, spk, p, bar);
  snn_out_kernel<<<BB, 256, 0, stream>>>(spk, w_out, (float*)d_out);
}
